// ParamTable_17712445129393
// MI455X (gfx1250) — matrix operand order repaired and checked by reading, not compiled, not run on hardware
//
#include <hip/hip_runtime.h>

typedef __attribute__((ext_vector_type(16))) _Float16 v16h;
typedef __attribute__((ext_vector_type(8)))  float    v8f;

// One wave (32 lanes) handles 32 batch elements via a single
// V_WMMA_F32_16X16X32_F16 that computes the base-2 row encodings exactly.
__global__ __launch_bounds__(256) void paramtable_wmma_kernel(
    const int* __restrict__ x_pa,     // [B,16] int32 in {0,1}
    const float* __restrict__ params, // [65536,2] float32
    float* __restrict__ out,          // [2*B]: out[0:B]=col0, out[B:2B]=col1
    long long Btotal)
{
  const int lane = (int)(threadIdx.x & 31u);
  const int waveInBlock = (int)(threadIdx.x >> 5);
  const long long waveBase =
      ((long long)blockIdx.x * 8 + waveInBlock) * 32;  // first element of this wave

  // Warm L2 with the 512KB gather table (emits global_prefetch_b8).
  {
    long long gid = (long long)blockIdx.x * blockDim.x + threadIdx.x;
    if (gid < (1LL << 17)) {  // 131072 floats = whole params table
      __builtin_prefetch(params + gid, 0, 3);
    }
  }

  const int m    = lane & 15;   // matrix row this lane contributes to
  const int half = lane >> 4;   // 0: K 0-15, 1: K 16-31

  // ---- Build A: row m = elements (2m, 2m+1); lane half h supplies all 16
  // states of element 2m+h (contiguous K = 16h + s).
  const long long e0 = waveBase + 2 * (long long)m;
  const int* p0 = x_pa + e0 * 16 + half * 16;

  int4 u0 = *(const int4*)(p0 + 0);   // element 2m+half, states 0..3
  int4 u1 = *(const int4*)(p0 + 4);   // element 2m+half, states 4..7
  int4 u2 = *(const int4*)(p0 + 8);   // element 2m+half, states 8..11
  int4 u3 = *(const int4*)(p0 + 12);  // element 2m+half, states 12..15

  v16h a;
  a[0]  = (_Float16)u0.x;  a[1]  = (_Float16)u0.y;
  a[2]  = (_Float16)u0.z;  a[3]  = (_Float16)u0.w;
  a[4]  = (_Float16)u1.x;  a[5]  = (_Float16)u1.y;
  a[6]  = (_Float16)u1.z;  a[7]  = (_Float16)u1.w;
  a[8]  = (_Float16)u2.x;  a[9]  = (_Float16)u2.y;
  a[10] = (_Float16)u2.z;  a[11] = (_Float16)u2.w;
  a[12] = (_Float16)u3.x;  a[13] = (_Float16)u3.y;
  a[14] = (_Float16)u3.z;  a[15] = (_Float16)u3.w;

  // ---- Build B: B[k][0] = 2^(15-k) for k<16, B[k][1] = 2^(15-(k-16)) for
  // k>=16, all other columns zero. In this code's contiguous B K order (lane = N for
  // lanes 0-15 with K 0-15, lane-16 = N for lanes 16-31 with K 16-31), the
  // nonzero values live only in lanes where (N == K-group), i.e. lanes 0 and 17.
  const bool bnz = (m == half);
  v16h b;
#pragma unroll
  for (int i = 0; i < 16; ++i) {
    // weight for the i-th K within this lane's K-group: 2^(15-i), exact in f16
    b[i] = bnz ? (_Float16)(float)(1 << (15 - i)) : (_Float16)0.0f;
  }

  v8f c = {0.f, 0.f, 0.f, 0.f, 0.f, 0.f, 0.f, 0.f};
  c = __builtin_amdgcn_wmma_f32_16x16x32_f16(
      /*neg_a=*/false, a, /*neg_b=*/false, b,
      /*c_mod=*/(short)0, c, /*reuse_a=*/false, /*reuse_b=*/false);

  // ---- Redistribute: lane t wants idx(element waveBase+t) = D[t/2][t%2],
  // which sits in VGPR ((t>>1)&7) of lane ((t&16)|(t&1)).
  const int src  = (lane & 16) | (lane & 1);
  const int rsel = (lane >> 1) & 7;
  float fidx = 0.0f;
#pragma unroll
  for (int r = 0; r < 8; ++r) {
    int vi = __builtin_amdgcn_ds_bpermute(src << 2, __float_as_int(c[r]));
    float v = __int_as_float(vi);
    if (r == rsel) fidx = v;
  }
  int idx = (int)(fidx + 0.5f);  // exact integer in [0, 65535]

  // ---- Gather + split-column store (both output streams coalesced).
  const long long e = waveBase + lane;
  float2 pr = *(const float2*)(params + 2 * (long long)idx);
  out[e]          = pr.x;
  out[Btotal + e] = pr.y;
}

// Scalar Horner tail for B not divisible by 256 (unused for B = 2^21).
__global__ void paramtable_tail_kernel(
    const int* __restrict__ x_pa, const float* __restrict__ params,
    float* __restrict__ out, long long start, long long Btotal)
{
  long long e = start + (long long)blockIdx.x * blockDim.x + threadIdx.x;
  if (e >= Btotal) return;
  int idx = 0;
#pragma unroll
  for (int j = 0; j < 16; ++j) idx = (idx << 1) + x_pa[e * 16 + j];
  out[e]          = params[2 * idx];
  out[Btotal + e] = params[2 * idx + 1];
}

extern "C" void kernel_launch(void* const* d_in, const int* in_sizes, int n_in,
                              void* d_out, int out_size, void* d_ws, size_t ws_size,
                              hipStream_t stream) {
  // inputs (setup_inputs order): x [B] int32 (unused), x_pa [B,16] int32,
  // params [65536,2] float32
  const int*   x_pa   = (const int*)d_in[1];
  const float* params = (const float*)d_in[2];
  float*       out    = (float*)d_out;

  const long long Btotal = (long long)in_sizes[0];
  const long long mainElems = (Btotal / 256) * 256;

  if (mainElems > 0) {
    dim3 grid((unsigned)(mainElems / 256)), block(256);
    hipLaunchKernelGGL(paramtable_wmma_kernel, grid, block, 0, stream,
                       x_pa, params, out, Btotal);
  }
  const long long tail = Btotal - mainElems;
  if (tail > 0) {
    dim3 grid((unsigned)((tail + 255) / 256)), block(256);
    hipLaunchKernelGGL(paramtable_tail_kernel, grid, block, 0, stream,
                       x_pa, params, out, mainElems, Btotal);
  }
}
